// DGCNN_60387240181995
// MI455X (gfx1250) — matrix operand order repaired and checked by reading, not compiled, not run on hardware
//
#include <hip/hip_runtime.h>
#include <hip/hip_bf16.h>

// ---------------------------------------------------------------------------
// DGCNN forward for MI455X (gfx1250): bf16 WMMA GEMM cores, fused BN stats.
// GEMM: 32x32 output block per wave (4 accumulators), K-loop software-
// pipelined 2-deep so fragment loads for step k+32 issue before the WMMAs of
// step k -- the s_wait before each WMMA group covers only stage-old loads.
// ---------------------------------------------------------------------------

#define BATCH 8
#define NPTS  1024
#define DIMS  60
#define KNN   20
#define M2D   (BATCH * NPTS * KNN)   // 163840 rows for the 2D conv GEMMs
#define M1D   (BATCH * NPTS)         // 8192 rows for the 1D conv GEMMs
#define EPS   1e-5f
#define SLOPE 0.2f

typedef __attribute__((ext_vector_type(16))) __bf16 v16bf;
typedef __attribute__((ext_vector_type(8)))  __bf16 v8bf;
typedef __attribute__((ext_vector_type(8)))  float  v8f;

// ---------------------------------------------------------------------------
// 1) kNN: per point, squared distances against LDS-tiled point set, top-20
//    (pd = -||xi - xj||^2, keep largest; self-distance 0 is the max -> first)
// ---------------------------------------------------------------------------
__global__ void __launch_bounds__(256) knn_kernel(const float* __restrict__ x,
                                                  int* __restrict__ idx) {
  const int b = blockIdx.y;
  const int n = blockIdx.x * 256 + threadIdx.x;
  const float* xb = x + (size_t)b * DIMS * NPTS;

  float xf[DIMS];
#pragma unroll
  for (int d = 0; d < DIMS; ++d) xf[d] = xb[d * NPTS + n];

  float best[KNN];
  int   bidx[KNN];
#pragma unroll
  for (int i = 0; i < KNN; ++i) { best[i] = -3.0e38f; bidx[i] = 0; }

  __shared__ float xs[DIMS][256];

  for (int mt = 0; mt < NPTS / 256; ++mt) {
    __syncthreads();
    for (int i = threadIdx.x; i < DIMS * 256; i += 256) {
      int d = i >> 8, mm = i & 255;
      xs[d][mm] = xb[d * NPTS + mt * 256 + mm];
    }
    __syncthreads();
    for (int mm = 0; mm < 256; ++mm) {
      float acc = 0.f;
#pragma unroll
      for (int d = 0; d < DIMS; ++d) {
        float df = xf[d] - xs[d][mm];
        acc = fmaf(df, df, acc);
      }
      float pd = -acc;
      if (pd > best[KNN - 1]) {
        int p = KNN - 1;
        while (p > 0 && best[p - 1] < pd) {  // strict: ties keep lower index first
          best[p] = best[p - 1]; bidx[p] = bidx[p - 1]; --p;
        }
        best[p] = pd; bidx[p] = mt * 256 + mm;
      }
    }
  }

  int* orow = idx + ((size_t)b * NPTS + n) * KNN;
#pragma unroll
  for (int i = 0; i < KNN; ++i) orow[i] = bidx[i];
}

// ---------------------------------------------------------------------------
// 2) Edge features -> bf16 A-matrix for conv1: rows r=((b*N+n)*k+j), 128 cols
//    c<60: x[b,c,idx]-x[b,c,n] ; 60<=c<120: x[b,c-60,n] ; c>=120: 0 (K pad)
// ---------------------------------------------------------------------------
__global__ void __launch_bounds__(256) edge_feat_kernel(const float* __restrict__ x,
                                                        const int* __restrict__ idx,
                                                        __bf16* __restrict__ A1) {
  size_t gid = (size_t)blockIdx.x * 256 + threadIdx.x;  // over M2D*128
  int    c   = (int)(gid & 127);
  size_t r   = gid >> 7;
  size_t bn  = r / KNN;
  int    n   = (int)(bn & (NPTS - 1));
  int    b   = (int)(bn >> 10);
  float v;
  if (c < DIMS) {
    int m = idx[r];
    const float* row = x + ((size_t)b * DIMS + c) * NPTS;
    v = row[m] - row[n];
  } else if (c < 2 * DIMS) {
    v = x[((size_t)b * DIMS + (c - DIMS)) * NPTS + n];
  } else {
    v = 0.f;
  }
  A1[gid] = (__bf16)v;
}

// ---------------------------------------------------------------------------
// 3) Weight convert: w (O x C f32) -> Wb (O x Kpad bf16), zero-padded K
// ---------------------------------------------------------------------------
__global__ void __launch_bounds__(256) wconv_kernel(const float* __restrict__ w,
                                                    __bf16* __restrict__ wb,
                                                    int O, int C, int Kpad) {
  int gid = blockIdx.x * 256 + threadIdx.x;
  if (gid >= O * Kpad) return;
  int o = gid / Kpad, k = gid % Kpad;
  wb[gid] = (k < C) ? (__bf16)w[o * C + k] : (__bf16)0.f;
}

__global__ void __launch_bounds__(256) zero_kernel(float* __restrict__ p, int n) {
  int g = blockIdx.x * 256 + threadIdx.x;
  if (g < n) p[g] = 0.f;
}

// ---------------------------------------------------------------------------
// Fragment loaders per CDNA5 ISA 7.12.2 VGPR layouts (bf16, 16x16x32):
//   A lane L (row m=L&15): K = {kk + 8*(L>=16) + 0..7, same +16}
//   B lane L (col n=L&15): K = {kk + 8*(L>=16) + 0..7, same +16}
// Both are two contiguous 16B loads from row-major [rows x Kpad] storage.
// ---------------------------------------------------------------------------
__device__ __forceinline__ v16bf ldfragA(const __bf16* row, int kk, int half) {
  union { v16bf v; v8bf h[2]; } u;
  u.h[0] = *(const v8bf*)(row + kk + half * 8);
  u.h[1] = *(const v8bf*)(row + kk + 16 + half * 8);
  return u.v;
}
__device__ __forceinline__ v16bf ldfragB(const __bf16* row, int kk, int half) {
  union { v16bf v; v8bf h[2]; } u;
  u.h[0] = *(const v8bf*)(row + kk + half * 8);
  u.h[1] = *(const v8bf*)(row + kk + 16 + half * 8);
  return u.v;
}
#define WMMA_BF16(a, b, c)                                                \
  __builtin_amdgcn_wmma_f32_16x16x32_bf16(false, (a), false, (b),         \
                                          (short)0, (c), false, false)

// ---------------------------------------------------------------------------
// 4) WMMA GEMM + bias, fused per-channel sum / sumsq (BN stats) via atomics.
//    A: [M x Kpad] bf16 row-major, W: [O x Kpad] bf16 row-major (B columns),
//    Y: [M x O] f32. Requires Kpad % 64 == 0 (true for all layers).
//    C/D layout: lane L, vgpr v -> m = mbase + v + 8*(L>=16), n = nbase+(L&15)
// ---------------------------------------------------------------------------
__global__ void __launch_bounds__(256)
gemm_bn_kernel(const __bf16* __restrict__ A, const __bf16* __restrict__ W,
               const float* __restrict__ bias, float* __restrict__ Y,
               float* __restrict__ stats, int M, int Kpad, int O) {
  const int  wave = threadIdx.x >> 5;
  const int  lane = threadIdx.x & 31;
  const int  tiles_o = O >> 5;                       // 32-wide N blocks
  const long tileId  = (long)blockIdx.x * 8 + wave;
  const long nTiles  = (long)(M >> 5) * tiles_o;
  if (tileId >= nTiles) return;  // wave-uniform: EXEC stays all-ones for WMMA

  const int  o0   = (int)(tileId % tiles_o) << 5;
  const int  m0   = (int)((tileId / tiles_o) << 5);
  const int  half = lane >> 4;   // 0: lanes 0-15, 1: lanes 16-31
  const int  l16  = lane & 15;

  const __bf16* arow0 = A + (size_t)(m0 + l16) * Kpad;
  const __bf16* arow1 = arow0 + (size_t)16 * Kpad;
  const __bf16* brow0 = W + (size_t)(o0 + l16) * Kpad;
  const __bf16* brow1 = brow0 + (size_t)16 * Kpad;

  v8f acc00 = {}, acc01 = {}, acc10 = {}, acc11 = {};

  // Stage 0 preload (kk = 0)
  v16bf ca0 = ldfragA(arow0, 0, half);
  v16bf ca1 = ldfragA(arow1, 0, half);
  v16bf cb0 = ldfragB(brow0, 0, half);
  v16bf cb1 = ldfragB(brow1, 0, half);

  for (int kk = 0; kk < Kpad; kk += 64) {
    // Issue loads for stage kk+32 before consuming stage kk.
    v16bf na0 = ldfragA(arow0, kk + 32, half);
    v16bf na1 = ldfragA(arow1, kk + 32, half);
    v16bf nb0 = ldfragB(brow0, kk + 32, half);
    v16bf nb1 = ldfragB(brow1, kk + 32, half);

    acc00 = WMMA_BF16(ca0, cb0, acc00);
    acc01 = WMMA_BF16(ca0, cb1, acc01);
    acc10 = WMMA_BF16(ca1, cb0, acc10);
    acc11 = WMMA_BF16(ca1, cb1, acc11);

    if (kk + 64 < Kpad) {
      // Loads for stage kk+64 overlap the WMMAs on stage kk+32.
      __builtin_prefetch(arow0 + kk + 96, 0, 3);  // global_prefetch_b8
      __builtin_prefetch(brow0 + kk + 96, 0, 3);
      ca0 = ldfragA(arow0, kk + 64, half);
      ca1 = ldfragA(arow1, kk + 64, half);
      cb0 = ldfragB(brow0, kk + 64, half);
      cb1 = ldfragB(brow1, kk + 64, half);
    }

    acc00 = WMMA_BF16(na0, nb0, acc00);
    acc01 = WMMA_BF16(na0, nb1, acc01);
    acc10 = WMMA_BF16(na1, nb0, acc10);
    acc11 = WMMA_BF16(na1, nb1, acc11);
  }

  // Epilogue: bias add, store 32x32 f32 block, per-channel sum/sumsq atomics.
  v8f accs[2][2] = {{acc00, acc01}, {acc10, acc11}};
#pragma unroll
  for (int j = 0; j < 2; ++j) {
    const int   n  = o0 + j * 16 + l16;
    const float bv = bias[n];
    float s = 0.f, sq = 0.f;
#pragma unroll
    for (int i = 0; i < 2; ++i) {
      const int mbase = m0 + i * 16 + half * 8;
#pragma unroll
      for (int v = 0; v < 8; ++v) {
        float d = accs[i][j][v] + bv;
        Y[(size_t)(mbase + v) * O + n] = d;
        s += d;
        sq = fmaf(d, d, sq);
      }
    }
    atomicAdd(&stats[n], s);          // per-channel sum
    atomicAdd(&stats[1024 + n], sq);  // per-channel sum of squares
  }
}

// ---------------------------------------------------------------------------
// 5) BN + leakyReLU + (bf16 A for next conv) + max over k -> Xcat column block
// ---------------------------------------------------------------------------
__global__ void __launch_bounds__(256)
bnact2d_kernel(const float* __restrict__ Y, const float* __restrict__ stats,
               const float* __restrict__ g, const float* __restrict__ be,
               __bf16* __restrict__ Anext, __bf16* __restrict__ Xcat,
               int O, int colOff, float invM) {
  int gid = blockIdx.x * 256 + threadIdx.x;  // over M1D * O
  if (gid >= M1D * O) return;
  int c  = gid % O;
  int bn = gid / O;
  float mean = stats[c] * invM;
  float var  = stats[1024 + c] * invM - mean * mean;
  float sc   = rsqrtf(var + EPS) * g[c];
  float sh   = be[c];

  const float* yrow = Y + (size_t)bn * KNN * O + c;
  __bf16* aout = Anext ? Anext + (size_t)bn * KNN * O + c : (__bf16*)0;
  float mx = -3.0e38f;
#pragma unroll 4
  for (int j = 0; j < KNN; ++j) {
    float v = (yrow[(size_t)j * O] - mean) * sc + sh;
    v = (v >= 0.f) ? v : SLOPE * v;
    if (aout) aout[(size_t)j * O] = (__bf16)v;
    mx = fmaxf(mx, v);
  }
  Xcat[(size_t)bn * 512 + colOff + c] = (__bf16)mx;
}

// ---------------------------------------------------------------------------
// 6) 1D BN + leakyReLU -> bf16 A (conv6 input) or f32 final output (B,N,1024)
// ---------------------------------------------------------------------------
__global__ void __launch_bounds__(256)
bnact1d_kernel(const float* __restrict__ Y, const float* __restrict__ stats,
               const float* __restrict__ g, const float* __restrict__ be,
               __bf16* __restrict__ outBf, float* __restrict__ outF,
               int O, float invM) {
  int gid = blockIdx.x * 256 + threadIdx.x;  // over M1D * O
  if (gid >= M1D * O) return;
  int c = gid % O;
  float mean = stats[c] * invM;
  float var  = stats[1024 + c] * invM - mean * mean;
  float v = (Y[gid] - mean) * rsqrtf(var + EPS) * g[c] + be[c];
  v = (v >= 0.f) ? v : SLOPE * v;
  if (outBf) outBf[gid] = (__bf16)v;
  if (outF)  outF[gid]  = v;
}

// ---------------------------------------------------------------------------
// Host-side orchestration
// ---------------------------------------------------------------------------
static inline int cdiv_i(long a, int b) { return (int)((a + b - 1) / b); }

extern "C" void kernel_launch(void* const* d_in, const int* in_sizes, int n_in,
                              void* d_out, int out_size, void* d_ws, size_t ws_size,
                              hipStream_t stream) {
  (void)in_sizes; (void)n_in; (void)out_size; (void)ws_size;

  const float* x = (const float*)d_in[0];
  const float *W[6], *Bi[6], *G[6], *Be[6];
  for (int i = 0; i < 6; ++i) {
    W[i]  = (const float*)d_in[1 + 4 * i];
    Bi[i] = (const float*)d_in[2 + 4 * i];
    G[i]  = (const float*)d_in[3 + 4 * i];
    Be[i] = (const float*)d_in[4 + 4 * i];
  }
  float* out = (float*)d_out;

  struct LW { int O, C, Kpad; };
  const LW lw[6] = {{64,120,128},{64,64,64},{128,64,64},
                    {256,128,128},{256,512,512},{1024,256,256}};

  // Workspace carve-up (256B aligned)
  char* p = (char*)d_ws;
  auto alloc = [&](size_t bytes) -> char* {
    char* r = p; p += (bytes + 255) & ~(size_t)255; return r;
  };
  int*    idx   = (int*)alloc((size_t)BATCH * NPTS * KNN * 4);
  float*  stats = (float*)alloc(2048 * 4);
  __bf16* wb[6];
  for (int i = 0; i < 6; ++i)
    wb[i] = (__bf16*)alloc((size_t)lw[i].O * lw[i].Kpad * 2);
  __bf16* bufA = (__bf16*)alloc((size_t)M2D * 128 * 2);   // ping
  __bf16* bufB = (__bf16*)alloc((size_t)M2D * 128 * 2);   // pong
  float*  Y    = (float*)alloc((size_t)M2D * 256 * 4);    // largest f32 pre-BN
  __bf16* xcat = (__bf16*)alloc((size_t)M1D * 512 * 2);   // concat(x1..x4)

  // --- graph + edge features ---
  knn_kernel<<<dim3(NPTS / 256, BATCH), 256, 0, stream>>>(x, idx);
  edge_feat_kernel<<<cdiv_i((long)M2D * 128, 256), 256, 0, stream>>>(x, idx, bufA);
  for (int i = 0; i < 6; ++i)
    wconv_kernel<<<cdiv_i((long)lw[i].O * lw[i].Kpad, 256), 256, 0, stream>>>(
        W[i], wb[i], lw[i].O, lw[i].C, lw[i].Kpad);

  auto run_gemm = [&](const __bf16* A, int li, int M) {
    zero_kernel<<<8, 256, 0, stream>>>(stats, 2048);
    long nT = (long)(M >> 5) * (lw[li].O >> 5);  // 32x32 blocks
    gemm_bn_kernel<<<cdiv_i(nT, 8), 256, 0, stream>>>(
        A, wb[li], Bi[li], Y, stats, M, lw[li].Kpad, lw[li].O);
  };

  const float inv2d = 1.0f / (float)M2D;
  const float inv1d = 1.0f / (float)M1D;

  // conv1 (120->64), BN, lrelu, max -> xcat[:,0:64); bf16 A2 -> bufB
  run_gemm(bufA, 0, M2D);
  bnact2d_kernel<<<cdiv_i((long)M1D * 64, 256), 256, 0, stream>>>(
      Y, stats, G[0], Be[0], bufB, xcat, 64, 0, inv2d);
  // conv2 (64->64) -> xcat[:,64:128); A3 -> bufA
  run_gemm(bufB, 1, M2D);
  bnact2d_kernel<<<cdiv_i((long)M1D * 64, 256), 256, 0, stream>>>(
      Y, stats, G[1], Be[1], bufA, xcat, 64, 64, inv2d);
  // conv3 (64->128) -> xcat[:,128:256); A4 -> bufB
  run_gemm(bufA, 2, M2D);
  bnact2d_kernel<<<cdiv_i((long)M1D * 128, 256), 256, 0, stream>>>(
      Y, stats, G[2], Be[2], bufB, xcat, 128, 128, inv2d);
  // conv4 (128->256) -> xcat[:,256:512); no next 2D A
  run_gemm(bufB, 3, M2D);
  bnact2d_kernel<<<cdiv_i((long)M1D * 256, 256), 256, 0, stream>>>(
      Y, stats, G[3], Be[3], (__bf16*)0, xcat, 256, 256, inv2d);
  // conv5 (512->256) on xcat; bf16 A6 -> bufA
  run_gemm(xcat, 4, M1D);
  bnact1d_kernel<<<cdiv_i((long)M1D * 256, 256), 256, 0, stream>>>(
      Y, stats, G[4], Be[4], bufA, (float*)0, 256, inv1d);
  // conv6 (256->1024); final BN+lrelu straight into (B,N,1024) f32 output
  run_gemm(bufA, 5, M1D);
  bnact1d_kernel<<<cdiv_i((long)M1D * 1024, 256), 256, 0, stream>>>(
      Y, stats, G[5], Be[5], (__bf16*)0, out, 1024, inv1d);
}
